// WindowAttention_4930622456224
// MI455X (gfx1250) — compile-verified, matrix operand order checked by reading, not run on hardware
//
#include <hip/hip_runtime.h>

#define NTOK 49
#define NP   64
#define DIM  128
#define HEADS 4
#define HD   32
#define QSCALE 0.17677669529663687f  // 32^-0.5

typedef __attribute__((ext_vector_type(16))) _Float16 v16h;
typedef __attribute__((ext_vector_type(8)))  float    v8f;

__device__ __forceinline__ int frag_k(int v, int lane) {
  // K offset for element-pair v (0..7) per ISA 16-bit A/B layout (16x16x32)
  return ((v >> 2) << 4) + ((lane >> 4) << 3) + ((v & 3) << 1);
}

// A-matrix from row-major [M][K] LDS, or B = T^T from row-major [N][K] LDS:
// per-lane major index = maj0 + (lane&15), contiguous along K.
__device__ __forceinline__ v16h frag_contig_lds(const _Float16* p, int stride,
                                                int maj0, int k0, int lane) {
  v16h f;
  const _Float16* row = p + (maj0 + (lane & 15)) * stride + k0;
#pragma unroll
  for (int v = 0; v < 8; ++v) {
    int k = frag_k(v, lane);
    f[2 * v]     = row[k];
    f[2 * v + 1] = row[k + 1];
  }
  return f;
}

// B-matrix from row-major [K][N] LDS storage.
__device__ __forceinline__ v16h frag_strided_lds(const _Float16* p, int stride,
                                                 int k0, int n0, int lane) {
  v16h f;
  const _Float16* col = p + n0 + (lane & 15);
#pragma unroll
  for (int v = 0; v < 8; ++v) {
    int k = frag_k(v, lane);
    f[2 * v]     = col[(k0 + k) * stride];
    f[2 * v + 1] = col[(k0 + k + 1) * stride];
  }
  return f;
}

// B = W^T fragment loaded straight from global row-major W[N][K] (f32 -> f16).
__device__ __forceinline__ v16h frag_wT_global(const float* __restrict__ W, int stride,
                                               int n0, int k0, int lane) {
  v16h f;
  const float* row = W + (n0 + (lane & 15)) * stride + k0;
#pragma unroll
  for (int v = 0; v < 8; ++v) {
    int k = frag_k(v, lane);
    f[2 * v]     = (_Float16)row[k];
    f[2 * v + 1] = (_Float16)row[k + 1];
  }
  return f;
}

__device__ __forceinline__ v8f wmma_f32_f16(v16h a, v16h b, v8f c) {
  return __builtin_amdgcn_wmma_f32_16x16x32_f16(false, a, false, b, (short)0, c,
                                                false, false);
}

__global__ __launch_bounds__(128, 2)
void swin_window_attn(const float* __restrict__ x,
                      const float* __restrict__ qkv_w,
                      const float* __restrict__ qkv_b,
                      const float* __restrict__ proj_w,
                      const float* __restrict__ proj_b,
                      const float* __restrict__ bias_table,
                      const int*   __restrict__ rel_index,
                      float* __restrict__ out) {
  // 160 KB LDS, phase-aliased (2 blocks fill the 320 KB WGP):
  //   [  0K, 64K)  sS    f32 [4][64][64]  (scores; first 16K overlays dead sX)
  //   [  0K, 16K)  sX    f16 [64][128]    (x tile, dead after phase 2)
  //   [ 64K,112K)  sQKV  f16 [3][4][64][32]
  //   [ 64K, 89K)  stage f32 raw x tile   (async staging, dead before sQKV written)
  //   [112K,144K)  sP    f16 [4][64][64]  (softmax probs)
  //   [144K,160K)  sO    f16 [64][128]    (attention output)
  __shared__ __align__(16) unsigned char smem[160 * 1024];
  float*    sS    = (float*)smem;
  _Float16* sX    = (_Float16*)smem;
  _Float16* sQKV  = (_Float16*)(smem + 64 * 1024);
  const float* stage = (const float*)(smem + 64 * 1024);
  _Float16* sP    = (_Float16*)(smem + 112 * 1024);
  _Float16* sO    = (_Float16*)(smem + 144 * 1024);

  const int b    = blockIdx.x;
  const int tid  = threadIdx.x;
  const int lane = tid & 31;
  const int wave = tid >> 5;
  const int rhalf = (lane >> 4) << 3;  // +8 row offset for upper half-wave (C/D layout)
  const int m0w  = wave * 16;          // this wave's fixed M-tile origin

  const float* xb = x + (size_t)b * NTOK * DIM;

  // ---- Phase 1a: async-copy raw x tile (f32) into LDS staging ----
  // GLOBAL_LOAD_ASYNC_TO_LDS_B128, tracked by ASYNCcnt (no VGPR round-trip).
  {
    const int nvec = (NTOK * DIM * 4) / 16;  // 1568 x 16B
    for (int i = tid; i < nvec; i += 128) {
      unsigned ldsoff = (unsigned)(64 * 1024 + i * 16);
      unsigned long long ga =
          (unsigned long long)(const void*)xb + (unsigned long long)i * 16;
      asm volatile("global_load_async_to_lds_b128 %0, %1, off"
                   :: "v"(ldsoff), "v"(ga) : "memory");
    }
    asm volatile("s_wait_asynccnt 0" ::: "memory");
  }
  __syncthreads();

  // ---- Phase 1b: convert staged x f32 -> f16 into sX, zero-pad rows 49..63 ----
  for (int i = tid * 4; i < NTOK * DIM; i += 128 * 4) {
    float4 v4 = *(const float4*)(stage + i);
    sX[i + 0] = (_Float16)v4.x;
    sX[i + 1] = (_Float16)v4.y;
    sX[i + 2] = (_Float16)v4.z;
    sX[i + 3] = (_Float16)v4.w;
  }
  for (int i = NTOK * DIM + tid; i < NP * DIM; i += 128) sX[i] = (_Float16)0.f;
  __syncthreads();

  // ---- Phase 2: QKV = x @ qkv_w^T + qkv_b (Q pre-scaled), scatter per head ----
  // Wave owns M-tile row `wave`; A-fragments hoisted across all 24 N-tiles.
  {
    v16h aX[4];
#pragma unroll
    for (int kt = 0; kt < 4; ++kt)
      aX[kt] = frag_contig_lds(sX, DIM, m0w, kt * 32, lane);
    for (int nt = 0; nt < 24; ++nt) {
      int n0 = nt * 16;
      v8f acc = {};
#pragma unroll
      for (int kt = 0; kt < 4; ++kt)
        acc = wmma_f32_f16(aX[kt], frag_wT_global(qkv_w, DIM, n0, kt * 32, lane), acc);
      int col   = n0 + (lane & 15);
      int which = col >> 7;           // 0=q, 1=k, 2=v  (tiles never straddle)
      int h     = (col & 127) >> 5;
      int d     = col & 31;
      float bias = qkv_b[col];
      float scl  = (which == 0) ? QSCALE : 1.f;
      _Float16* dst = sQKV + (which * HEADS + h) * NP * HD + d;
#pragma unroll
      for (int r = 0; r < 8; ++r)
        dst[(m0w + r + rhalf) * HD] = (_Float16)((acc[r] + bias) * scl);
    }
  }
  __syncthreads();

  // ---- Phase 3: per-head scores S = Q @ K^T (wave = head) ----
  // All 4 K^T B-fragments hoisted; one A-fragment per M-tile.
  {
    const _Float16* Q = sQKV + (0 * HEADS + wave) * NP * HD;
    const _Float16* K = sQKV + (1 * HEADS + wave) * NP * HD;
    float* S = sS + wave * NP * NP;
    v16h bK[4];
#pragma unroll
    for (int nt = 0; nt < 4; ++nt)
      bK[nt] = frag_contig_lds(K, HD, nt * 16, 0, lane);  // B = K^T
#pragma unroll
    for (int mt = 0; mt < 4; ++mt) {
      v16h a = frag_contig_lds(Q, HD, mt * 16, 0, lane);
#pragma unroll
      for (int nt = 0; nt < 4; ++nt) {
        v8f acc = {};
        acc = wmma_f32_f16(a, bK[nt], acc);
        int colc = nt * 16 + (lane & 15);
#pragma unroll
        for (int r = 0; r < 8; ++r)
          S[(mt * 16 + r + rhalf) * NP + colc] = acc[r];
      }
    }
  }
  __syncthreads();

  // ---- Phase 4: softmax(S + rel_pos_bias) -> P (f16), lane-per-row ----
  {
    const int h = wave;
    float*    S = sS + h * NP * NP;
    _Float16* P = sP + h * NP * NP;
    for (int r = lane; r < NP; r += 32) {
      if (r < NTOK) {
        const int* ri = rel_index + r * NTOK;
        float mx = -1e30f;
        for (int j = 0; j < NTOK; ++j) {
          float t = S[r * NP + j] + bias_table[ri[j] * HEADS + h];
          S[r * NP + j] = t;
          mx = fmaxf(mx, t);
        }
        float sum = 0.f;
        for (int j = 0; j < NTOK; ++j) {
          float e = __expf(S[r * NP + j] - mx);
          S[r * NP + j] = e;
          sum += e;
        }
        float inv = 1.f / sum;
        for (int j = 0; j < NTOK; ++j) P[r * NP + j] = (_Float16)(S[r * NP + j] * inv);
        for (int j = NTOK; j < NP; ++j) P[r * NP + j] = (_Float16)0.f;
      } else {
        for (int j = 0; j < NP; ++j) P[r * NP + j] = (_Float16)0.f;
      }
    }
  }
  __syncthreads();

  // ---- Phase 5: O = P @ V per head; V B-fragments hoisted ----
  {
    const _Float16* P = sP + wave * NP * NP;
    const _Float16* V = sQKV + (2 * HEADS + wave) * NP * HD;
    v16h bV[2][2];  // [kt][nt]
#pragma unroll
    for (int kt = 0; kt < 2; ++kt)
#pragma unroll
      for (int nt = 0; nt < 2; ++nt)
        bV[kt][nt] = frag_strided_lds(V, HD, kt * 32, nt * 16, lane);
#pragma unroll
    for (int mt = 0; mt < 4; ++mt) {
      v16h a0 = frag_contig_lds(P, NP, mt * 16, 0, lane);
      v16h a1 = frag_contig_lds(P, NP, mt * 16, 32, lane);
#pragma unroll
      for (int nt = 0; nt < 2; ++nt) {
        v8f acc = {};
        acc = wmma_f32_f16(a0, bV[0][nt], acc);
        acc = wmma_f32_f16(a1, bV[1][nt], acc);
        int colc = wave * HD + nt * 16 + (lane & 15);
#pragma unroll
        for (int r = 0; r < 8; ++r)
          sO[(mt * 16 + r + rhalf) * DIM + colc] = (_Float16)acc[r];
      }
    }
  }
  __syncthreads();

  // ---- Phase 6: out = O @ proj_w^T + proj_b (f32 store, rows < 49) ----
  // Wave owns M-tile row `wave`; A-fragments hoisted across the 8 N-tiles.
  {
    float* outb = out + (size_t)b * NTOK * DIM;
    v16h aO[4];
#pragma unroll
    for (int kt = 0; kt < 4; ++kt)
      aO[kt] = frag_contig_lds(sO, DIM, m0w, kt * 32, lane);
    for (int nt = 0; nt < 8; ++nt) {
      int n0 = nt * 16;
      v8f acc = {};
#pragma unroll
      for (int kt = 0; kt < 4; ++kt)
        acc = wmma_f32_f16(aO[kt], frag_wT_global(proj_w, DIM, n0, kt * 32, lane), acc);
      int colc = n0 + (lane & 15);
      float pb = proj_b[colc];
#pragma unroll
      for (int r = 0; r < 8; ++r) {
        int row = m0w + r + rhalf;
        if (row < NTOK) outb[row * DIM + colc] = acc[r] + pb;
      }
    }
  }
}

extern "C" void kernel_launch(void* const* d_in, const int* in_sizes, int n_in,
                              void* d_out, int out_size, void* d_ws, size_t ws_size,
                              hipStream_t stream) {
  const float* x          = (const float*)d_in[0];
  const float* qkv_w      = (const float*)d_in[1];
  const float* qkv_b      = (const float*)d_in[2];
  const float* proj_w     = (const float*)d_in[3];
  const float* proj_b     = (const float*)d_in[4];
  const float* bias_table = (const float*)d_in[5];
  const int*   rel_index  = (const int*)d_in[6];
  float* out = (float*)d_out;

  swin_window_attn<<<dim3(4096), dim3(128), 0, stream>>>(
      x, qkv_w, qkv_b, proj_w, proj_b, bias_table, rel_index, out);
}
